// TemporalRegression_17961553232503
// MI455X (gfx1250) — matrix operand order repaired and checked by reading, not compiled, not run on hardware
//
#include <hip/hip_runtime.h>
#include <hip/hip_bf16.h>
#include <math.h>

typedef __attribute__((ext_vector_type(16))) _Float16 v16h;
typedef __attribute__((ext_vector_type(8)))  float    v8f;

#define BATCH  32
#define C_IN   12
#define C_HID  16
#define H_     495
#define W_     436
#define HW     (H_ * W_)          // 215820
#define NPIX   (BATCH * HW)       // 6,906,240  (divisible by 64 -> all pairs full)
#define NPAIRS (NPIX / 64)        // 107,910 wave iterations of 64 pixels

union V16 {
    v16h v;
    _Float16 h[16];
    int i[8];
};

// Single-instruction f32 ReLU (skip fmax canonicalization; inputs are finite).
__device__ __forceinline__ float relu_f32(float x) {
    float r;
    asm("v_max_num_f32 %0, %1, 0" : "=v"(r) : "v"(x));
    return r;
}

// Per 32-pixel tile (one wave), ZERO shuffles in the matrix path:
//   Layer 1: lane l's 12 loaded channels already sit in B layout -- lanes 0-15
//   carry K=0-11 of columns 0-15 (pixels 0-15), lanes 16-31 carry K=16-27 of
//   the same columns (pixels 16-31). Two A matrices (W1 active at K=0-11 vs
//   K=16-27) turn ONE B into both 16-pixel output tiles.
//   Layer 2: hidden values repacked purely in-lane; two A matrices (full W2 row
//   at K=0-15 vs K=16-31, table order) again produce both tiles from one B.
// Two independent tiles (64 pixels) per loop iteration give the scheduler
// material to fill the WMMA->VALU hazard slots and load-wait shadows.
__global__ __launch_bounds__(256) void mlp1x1_wmma_kernel(
    const float* __restrict__ x,
    const float* __restrict__ w1, const float* __restrict__ b1,
    const float* __restrict__ w2, const float* __restrict__ b2,
    const float* __restrict__ w3, const float* __restrict__ b3,
    float* __restrict__ out)
{
    const int lane   = threadIdx.x & 31;
    const int waveId = blockIdx.x * (blockDim.x >> 5) + (threadIdx.x >> 5);
    const int nWaves = gridDim.x * (blockDim.x >> 5);

    const int row = lane & 15;       // A-matrix row index held by this lane
    const int hi  = lane >> 4;       // 0: lanes 0-15, 1: lanes 16-31
    const int kb  = hi * 8;          // K/channel base for A and C/D layouts

    // ---- Layer-1 A matrices: W1/255 at K=0-11 (a1a) or K=16-27 (a1b).
    // A 16x32 f16 layout (contiguous): lane<16: h0-15 => K0-15
    //                                  lane>=16: h0-15 => K16-31
    v16h a1a, a1b;
    float w1partial = 0.0f;          // this lane's W1 row-slice sum (unscaled)
    #pragma unroll
    for (int j = 0; j < 16; ++j) {
        const int k = j;
        const float wraw = (k < C_IN) ? w1[row * C_IN + k] : 0.0f;
        w1partial += hi ? 0.0f : wraw;
        const float wv = wraw * (1.0f / 255.0f);
        a1a[j] = hi ? (_Float16)0.f : (_Float16)wv;
        a1b[j] = hi ? (_Float16)wv : (_Float16)0.f;  // a1b = half-swapped a1a
    }
    const float w1rowsum = w1partial + __shfl_xor(w1partial, 16, 32);

    // ---- Layer-2 A matrices: W2 row at K=0-15 (a2a) or K=16-31 (a2b), table order.
    v16h a2a, a2b;
    #pragma unroll
    for (int h = 0; h < 8; ++h) {
        const _Float16 wv = (_Float16)w2[row * C_HID + kb + h];
        a2a[h]     = wv;
        a2a[h + 8] = (_Float16)0.f;
        a2b[h]     = (_Float16)0.f;
        a2b[h + 8] = wv;
    }

    // ---- bias-preloaded C accumulators (C/D layout: VGPR r -> ch r+kb) ----
    // Layer-1 bias absorbs "-0.5" input shift: b1' = b1 - 0.5 * sum_c W1[o][c]
    v8f cb1, cb2;
    float w3v[8];
    #pragma unroll
    for (int r = 0; r < 8; ++r) {
        const int o = r + kb;
        cb1[r] = b1[o] - 0.5f * __shfl(w1rowsum, o, 32);
        cb2[r] = b2[o];
        w3v[r] = w3[o];
    }
    const float bias3 = b3[0];
    const v16h zero16 = {};

    for (int t = waveId; t < NPAIRS; t += nWaves) {
        // two independent 32-pixel streams: pixels t*64 + {0..31, 32..63}
        int p[2];
        const float* xb[2];
        #pragma unroll
        for (int u = 0; u < 2; ++u) {
            p[u] = t * 64 + u * 32 + lane;
            const int b = p[u] / HW;
            const int s = p[u] - b * HW;
            xb[u] = x + (size_t)b * (size_t)(C_IN * HW) + s;
        }

        // 12 fully-coalesced 128B loads per stream; lands in-lane == B layout.
        V16 Bin[2];
        #pragma unroll
        for (int u = 0; u < 2; ++u) {
            #pragma unroll
            for (int c = 0; c < C_IN; ++c)
                Bin[u].h[c] = (_Float16)xb[u][c * HW];   // x raw; scale in W1/b1
            #pragma unroll
            for (int c = C_IN; c < 16; ++c)
                Bin[u].h[c] = (_Float16)0.f;
        }

        // ---- layer 1: one B, two A -> both 16-pixel tiles (per stream) ----
        v8f d0[2], d1[2];
        #pragma unroll
        for (int u = 0; u < 2; ++u) {
            d0[u] = __builtin_amdgcn_wmma_f32_16x16x32_f16(
                false, a1a, false, Bin[u].v, (short)0, cb1, false, false);
            d1[u] = __builtin_amdgcn_wmma_f32_16x16x32_f16(
                false, a1b, false, Bin[u].v, (short)0, cb1, false, false);
        }

        // ---- cvt to f16 then packed-f16 ReLU (8 cvt_pk + 8 pk_max per stream)
        V16 B2[2];
        #pragma unroll
        for (int u = 0; u < 2; ++u) {
            #pragma unroll
            for (int r = 0; r < 8; ++r) {
                B2[u].h[r]     = (_Float16)d0[u][r];
                B2[u].h[8 + r] = (_Float16)d1[u][r];
            }
            B2[u].v = __builtin_elementwise_max(B2[u].v, zero16);
        }

        // ---- layer 2: one B, two A -> both tiles (per stream) ----
        v8f e0[2], e1[2];
        #pragma unroll
        for (int u = 0; u < 2; ++u) {
            e0[u] = __builtin_amdgcn_wmma_f32_16x16x32_f16(
                false, a2a, false, B2[u].v, (short)0, cb2, false, false);
            e1[u] = __builtin_amdgcn_wmma_f32_16x16x32_f16(
                false, a2b, false, B2[u].v, (short)0, cb2, false, false);
        }

        // ---- layer 3: z = w3 . relu(E) + b3 ; out = 255*sigmoid(z) ----
        #pragma unroll
        for (int u = 0; u < 2; ++u) {
            float acc0 = 0.f, acc1 = 0.f;
            #pragma unroll
            for (int r = 0; r < 8; ++r) {
                acc0 = fmaf(relu_f32(e0[u][r]), w3v[r], acc0);
                acc1 = fmaf(relu_f32(e1[u][r]), w3v[r], acc1);
            }
            acc0 += __shfl_xor(acc0, 16, 32);   // combine ch0-7 / ch8-15 halves
            acc1 += __shfl_xor(acc1, 16, 32);

            const float z   = (hi ? acc1 : acc0) + bias3;
            const float res = 255.0f / (1.0f + __expf(-z));
            out[p[u]] = res;                    // coalesced 128B store per wave
        }
    }
}

extern "C" void kernel_launch(void* const* d_in, const int* in_sizes, int n_in,
                              void* d_out, int out_size, void* d_ws, size_t ws_size,
                              hipStream_t stream) {
    (void)in_sizes; (void)n_in; (void)d_ws; (void)ws_size; (void)out_size;
    const float* x  = (const float*)d_in[0];
    const float* w1 = (const float*)d_in[1];
    const float* b1 = (const float*)d_in[2];
    const float* w2 = (const float*)d_in[3];
    const float* b2 = (const float*)d_in[4];
    const float* w3 = (const float*)d_in[5];
    const float* b3 = (const float*)d_in[6];
    float* out = (float*)d_out;

    dim3 block(256);           // 8 wave32 waves per block
    dim3 grid(2048);           // 16384 waves, ~6.6 pair-iterations each
    mlp1x1_wmma_kernel<<<grid, block, 0, stream>>>(x, w1, b1, w2, b2, w3, b3, out);
}
